// FCOS_23553600651301
// MI455X (gfx1250) — matrix operand order repaired and checked by reading, not compiled, not run on hardware
//
#include <hip/hip_runtime.h>
#include <hip/hip_bf16.h>

typedef __attribute__((ext_vector_type(16))) _Float16 v16h;
typedef __attribute__((ext_vector_type(8)))  _Float16 v8h;
typedef __attribute__((ext_vector_type(8)))  float    v8f;

#define IC 256
#define GROUPS 32
#define CPG 8           // channels per group
#define TOT 20267       // sum of H*W over levels
#define NCLS 80
#define LDS_STRIDE 264  // f16 elements per LDS row (528B: 16B aligned, bank-skewed)

// d_out flat offsets (outputs concatenated in return order), per-output [2, TOT, c]
#define OFF_CLS   ((size_t)0)
#define OFF_BBOX  (OFF_CLS  + (size_t)2*TOT*80)
#define OFF_CTR   (OFF_BBOX + (size_t)2*TOT*4)
#define OFF_DIM   (OFF_CTR  + (size_t)2*TOT*1)
#define OFF_ORI   (OFF_DIM  + (size_t)2*TOT*3)
#define OFF_KP    (OFF_ORI  + (size_t)2*TOT*1)
#define OFF_DEPTH (OFF_KP   + (size_t)2*TOT*16)

// ---------------------------------------------------------------------------
// Pack f32 NCHW feature map -> f16 NHWC   (coalesced read, strided write)
// ---------------------------------------------------------------------------
__global__ void pack_in_kernel(const float* __restrict__ in, _Float16* __restrict__ out,
                               int HW, int total) {
    int idx = blockIdx.x * blockDim.x + threadIdx.x;
    if (idx >= total) return;
    int n   = idx / (IC * HW);
    int r   = idx - n * IC * HW;
    int c   = r / HW;
    int phw = r - c * HW;
    out[((size_t)(n * HW + phw)) * IC + c] = (_Float16)in[idx];
}

__global__ void zero_kernel(float* __restrict__ p, int n) {
    int idx = blockIdx.x * blockDim.x + threadIdx.x;
    if (idx < n) p[idx] = 0.f;
}

// ---------------------------------------------------------------------------
// Pack conv weights f32 [O, IC, 3, 3] -> f16 [kk][OCalloc][IC], zero-padded
// ---------------------------------------------------------------------------
__global__ void pack_w_kernel(const float* __restrict__ w, _Float16* __restrict__ wq,
                              int O, int OCalloc) {
    int idx = blockIdx.x * blockDim.x + threadIdx.x;
    int total = 9 * OCalloc * IC;
    if (idx >= total) return;
    int kk = idx / (OCalloc * IC);
    int r  = idx - kk * OCalloc * IC;
    int oc = r / IC;
    int ic = r - oc * IC;
    float v = (oc < O) ? w[((size_t)oc * IC + ic) * 9 + kk] : 0.f;
    wq[((size_t)kk * OCalloc + oc) * IC + ic] = (_Float16)v;
}

// Fused reg-head weights: [bbox0..3, ctr, dim0..2, ori, kp0..15, depth] -> 26, pad to 64
__global__ void pack_reg_head_kernel(const float* bw, const float* cw, const float* dw,
                                     const float* ow, const float* kw, const float* pw,
                                     _Float16* __restrict__ wq) {
    int idx = blockIdx.x * blockDim.x + threadIdx.x;
    const int OCalloc = 64;
    int total = 9 * OCalloc * IC;
    if (idx >= total) return;
    int kk = idx / (OCalloc * IC);
    int r  = idx - kk * OCalloc * IC;
    int oc = r / IC;
    int ic = r - oc * IC;
    float v = 0.f;
    if (oc < 4)        v = bw[((size_t)oc * IC + ic) * 9 + kk];
    else if (oc == 4)  v = cw[((size_t)ic) * 9 + kk];
    else if (oc < 8)   v = dw[((size_t)(oc - 5) * IC + ic) * 9 + kk];
    else if (oc == 8)  v = ow[((size_t)ic) * 9 + kk];
    else if (oc < 25)  v = kw[((size_t)(oc - 9) * IC + ic) * 9 + kk];
    else if (oc == 25) v = pw[((size_t)ic) * 9 + kk];
    wq[((size_t)kk * OCalloc + oc) * IC + ic] = (_Float16)v;
}

__global__ void pack_reg_bias_kernel(const float* bb, const float* cb, const float* db,
                                     const float* ob, const float* kb, const float* pb,
                                     float* __restrict__ bf) {
    int t = threadIdx.x;
    if (t >= 32) return;
    float v = 0.f;
    if (t < 4)        v = bb[t];
    else if (t == 4)  v = cb[0];
    else if (t < 8)   v = db[t - 5];
    else if (t == 8)  v = ob[0];
    else if (t < 25)  v = kb[t - 9];
    else if (t == 25) v = pb[0];
    bf[t] = v;
}

// ---------------------------------------------------------------------------
// Implicit-GEMM conv3x3 via V_WMMA_F32_16X16X32_F16 with double-buffered
// async global->LDS slab staging (global_load_async_to_lds_b128).
//   MODE 0: tower conv  -> f32 NHWC workspace (outT)
//   MODE 1: cls head    -> d_out cls region (realOC=80)
//   MODE 2: fused reg   -> d_out 6 regions (realOC=26, relu on bbox)
// block = 256 threads (8 waves) = 2 pixel-subtiles x 4 oc-subtiles
// block tile = 32 pixels x 64 output channels; K loop = 9 taps x 8 ic-blocks.
// Per tap: one 32x256 f16 slab staged async into LDS (ping-pong), 1 barrier.
// ---------------------------------------------------------------------------
template <int MODE>
__launch_bounds__(256)
__global__ void conv3x3_wmma_kernel(const _Float16* __restrict__ in,
                                    const _Float16* __restrict__ wq,
                                    const float* __restrict__ bias,
                                    float* __restrict__ outT,
                                    const float* __restrict__ zbuf,
                                    int H, int W, int HW, int numPix,
                                    int OCalloc, int realOC, int lvlOff,
                                    float* __restrict__ dout) {
    __shared__ _Float16 lsA[2][32][LDS_STRIDE];

    const int tid  = threadIdx.x;
    const int lane = tid & 31;
    const int wv   = tid >> 5;       // 0..7
    const int mi   = wv >> 2;        // pixel sub-tile 0..1
    const int oi   = wv & 3;         // oc sub-tile 0..3
    const int pixBase    = blockIdx.x * 32;
    const int pixSubBase = pixBase + mi * 16;
    const int ocBase     = blockIdx.y * 64 + oi * 16;

    // staging role: thread t -> slab row t/8 (32 rows), 64B column segment t%8
    const int srow = tid >> 3;
    const int sseg = tid & 7;
    const int sp   = pixBase + srow;
    const bool spv = sp < numPix;
    int sn = 0, sy = 0, sx = 0;
    if (spv) {
        sn = sp / HW;
        int rem = sp - sn * HW;
        sy = rem / W;
        sx = rem - sy * W;
    }
    const unsigned long long zaddr = (unsigned long long)(uintptr_t)zbuf;

    // issue async global->LDS copies for tap kk into buffer buf
    auto stage = [&](int kk, int buf) {
        const int dy = kk / 3 - 1, dx = kk % 3 - 1;
        const int iy = sy + dy, ix = sx + dx;
        const bool ok = spv && iy >= 0 && iy < H && ix >= 0 && ix < W;
        unsigned long long g = ok
            ? (unsigned long long)(uintptr_t)(in + (((size_t)sn * HW + (size_t)iy * W + ix) * IC + sseg * 32))
            : zaddr;
        unsigned l0 = (unsigned)(unsigned long long)(uintptr_t)&lsA[buf][srow][sseg * 32];
#pragma unroll
        for (int j = 0; j < 4; ++j) {
            unsigned laddr = l0 + j * 16;
            unsigned long long ga = g + (ok ? j * 16 : 0);
            asm volatile("global_load_async_to_lds_b128 %0, %1, off"
                         :: "v"(laddr), "v"(ga) : "memory");
        }
    };

    v8f acc = {};
    const int m  = lane & 15;
    const int k0 = (lane >> 4) * 16;

    stage(0, 0);

    for (int kk = 0; kk < 9; ++kk) {
        const int cur = kk & 1;
        asm volatile("s_wait_asynccnt 0" ::: "memory");
        __syncthreads();
        if (kk < 8) stage(kk + 1, cur ^ 1);

        const _Float16* arow = &lsA[cur][mi * 16 + m][0];
        const _Float16* bp =
            wq + (((size_t)kk * OCalloc + ocBase + (lane & 15)) * IC + (lane >> 4) * 16);

#pragma unroll
        for (int icb = 0; icb < 8; ++icb) {
            // A fragment: lanes 0-15 K{0..15}, lanes 16-31 K{16..31} (matches B)
            v8h c0 = *(const v8h*)(arow + icb * 32 + k0);
            v8h c1 = *(const v8h*)(arow + icb * 32 + k0 + 8);
            v16h a;
#pragma unroll
            for (int i = 0; i < 8; ++i) { a[i] = c0[i]; a[8 + i] = c1[i]; }

            // B fragment: 16 contiguous f16 (K-chunk) of column oc
            v16h b = *(const v16h*)(bp + (size_t)icb * 32);

            acc = __builtin_amdgcn_wmma_f32_16x16x32_f16(
                false, a, false, b, (short)0, acc, false, false);
        }
    }

    // ---- epilogue ----
    const int oc = ocBase + (lane & 15);
    const float bv = (oc < realOC) ? bias[oc] : 0.f;

    size_t hbase = 0; int hc = 1, hoff = 0; bool rl = false;
    if (MODE == 2 && oc < realOC) {
        if (oc < 4)        { hbase = OFF_BBOX;  hc = 4;  hoff = oc;      rl = true; }
        else if (oc == 4)  { hbase = OFF_CTR;   hc = 1;  hoff = 0; }
        else if (oc < 8)   { hbase = OFF_DIM;   hc = 3;  hoff = oc - 5; }
        else if (oc == 8)  { hbase = OFF_ORI;   hc = 1;  hoff = 0; }
        else if (oc < 25)  { hbase = OFF_KP;    hc = 16; hoff = oc - 9; }
        else               { hbase = OFF_DEPTH; hc = 1;  hoff = 0; }
    }

#pragma unroll
    for (int v = 0; v < 8; ++v) {
        const int mrow = v + ((lane >> 4) << 3);
        const int p = pixSubBase + mrow;
        if (p >= numPix || oc >= realOC) continue;
        float r = acc[v] + bv;
        if (MODE == 0) {
            outT[(size_t)p * IC + oc] = r;
        } else {
            const int n = p / HW;
            const int gp = lvlOff + (p - n * HW);
            if (MODE == 1) {
                dout[OFF_CLS + ((size_t)n * TOT + gp) * NCLS + oc] = r;
            } else {
                if (rl) r = fmaxf(r, 0.f);
                dout[hbase + ((size_t)n * TOT + gp) * hc + hoff] = r;
            }
        }
    }
}

// ---------------------------------------------------------------------------
// GroupNorm + ReLU + cvt->f16. One block per (n, group); deterministic tree
// reduction over 8 channels x HW elements of f32 NHWC input.
// ---------------------------------------------------------------------------
__launch_bounds__(256)
__global__ void gn_relu_kernel(const float* __restrict__ x,
                               const float* __restrict__ gw,
                               const float* __restrict__ gb,
                               _Float16* __restrict__ y, int HW) {
    __shared__ float ssum[256];
    __shared__ float ssq[256];
    __shared__ float s_mu, s_rinv;

    const int g = blockIdx.x & (GROUPS - 1);
    const int n = blockIdx.x >> 5;
    const int c0 = g * CPG;
    const int tid = threadIdx.x;
    const float* xp = x + (size_t)n * HW * IC;

    float s = 0.f, q = 0.f;
    for (int p = tid; p < HW; p += 256) {
        const float4* v4 = (const float4*)(xp + (size_t)p * IC + c0);
        float4 a = v4[0], b = v4[1];
        s += a.x + a.y + a.z + a.w + b.x + b.y + b.z + b.w;
        q += a.x * a.x + a.y * a.y + a.z * a.z + a.w * a.w +
             b.x * b.x + b.y * b.y + b.z * b.z + b.w * b.w;
    }
    ssum[tid] = s; ssq[tid] = q;
    __syncthreads();
    for (int st = 128; st > 0; st >>= 1) {
        if (tid < st) { ssum[tid] += ssum[tid + st]; ssq[tid] += ssq[tid + st]; }
        __syncthreads();
    }
    if (tid == 0) {
        float cnt = (float)(CPG * HW);
        float mu = ssum[0] / cnt;
        float var = fmaxf(ssq[0] / cnt - mu * mu, 0.f);
        s_mu = mu;
        s_rinv = rsqrtf(var + 1e-5f);
    }
    __syncthreads();
    const float mu = s_mu, rinv = s_rinv;

    float wv[CPG], bvv[CPG];
#pragma unroll
    for (int j = 0; j < CPG; ++j) { wv[j] = gw[c0 + j]; bvv[j] = gb[c0 + j]; }

    for (int p = tid; p < HW; p += 256) {
        const float* px = xp + (size_t)p * IC + c0;
        v8h o;
#pragma unroll
        for (int j = 0; j < CPG; ++j) {
            float v = (px[j] - mu) * rinv * wv[j] + bvv[j];
            v = fmaxf(v, 0.f);
            o[j] = (_Float16)v;
        }
        *(v8h*)(y + ((size_t)n * HW + p) * IC + c0) = o;
    }
}

// ---------------------------------------------------------------------------
extern "C" void kernel_launch(void* const* d_in, const int* in_sizes, int n_in,
                              void* d_out, int out_size, void* d_ws, size_t ws_size,
                              hipStream_t stream) {
    (void)in_sizes; (void)n_in; (void)out_size; (void)ws_size;

    static const int LH[5] = {100, 50, 25, 13, 7};
    static const int LW[5] = {152, 76, 38, 19, 10};
    static const int LOFF[5] = {0, 15200, 19000, 19950, 20197};

    const float* feats[5];
    for (int i = 0; i < 5; ++i) feats[i] = (const float*)d_in[i];
    const float* cls_conv_w = (const float*)d_in[5];
    const float* cls_conv_b = (const float*)d_in[6];
    const float* cls_gn_w   = (const float*)d_in[7];
    const float* cls_gn_b   = (const float*)d_in[8];
    const float* cls_out_w  = (const float*)d_in[9];
    const float* cls_out_b  = (const float*)d_in[10];
    const float* reg_conv_w = (const float*)d_in[11];
    const float* reg_conv_b = (const float*)d_in[12];
    const float* reg_gn_w   = (const float*)d_in[13];
    const float* reg_gn_b   = (const float*)d_in[14];
    const float* bbox_w = (const float*)d_in[15]; const float* bbox_b = (const float*)d_in[16];
    const float* ctr_w  = (const float*)d_in[17]; const float* ctr_b  = (const float*)d_in[18];
    const float* dim_w  = (const float*)d_in[19]; const float* dim_b  = (const float*)d_in[20];
    const float* ori_w  = (const float*)d_in[21]; const float* ori_b  = (const float*)d_in[22];
    const float* kp_w   = (const float*)d_in[23]; const float* kp_b   = (const float*)d_in[24];
    const float* dep_w  = (const float*)d_in[25]; const float* dep_b  = (const float*)d_in[26];

    // workspace carve-up
    char* ws = (char*)d_ws;
    size_t off = 0;
    auto carve = [&](size_t bytes) -> void* {
        off = (off + 255) & ~(size_t)255;
        void* p = ws + off;
        off += bytes;
        return p;
    };
    const size_t LAYW = (size_t)9 * 256 * IC * sizeof(_Float16);
    _Float16* wClsT = (_Float16*)carve(4 * LAYW);
    _Float16* wRegT = (_Float16*)carve(4 * LAYW);
    _Float16* wClsH = (_Float16*)carve((size_t)9 * 128 * IC * sizeof(_Float16));
    _Float16* wRegH = (_Float16*)carve((size_t)9 * 64 * IC * sizeof(_Float16));
    float*    bReg  = (float*)carve(32 * sizeof(float));
    float*    zbuf  = (float*)carve(256);
    const int maxPix = 2 * 15200;
    _Float16* actA  = (_Float16*)carve((size_t)maxPix * IC * sizeof(_Float16));
    _Float16* actB  = (_Float16*)carve((size_t)maxPix * IC * sizeof(_Float16));
    float*    convO = (float*)carve((size_t)maxPix * IC * sizeof(float));

    // --- pack weights + zero page (once per launch) ---
    {
        int tot = 9 * 256 * IC, blk = 256, grd = (tot + blk - 1) / blk;
        for (int i = 0; i < 4; ++i) {
            pack_w_kernel<<<grd, blk, 0, stream>>>(cls_conv_w + (size_t)i * 256 * IC * 9,
                                                   wClsT + (size_t)i * 9 * 256 * IC, 256, 256);
            pack_w_kernel<<<grd, blk, 0, stream>>>(reg_conv_w + (size_t)i * 256 * IC * 9,
                                                   wRegT + (size_t)i * 9 * 256 * IC, 256, 256);
        }
        int totH = 9 * 128 * IC;
        pack_w_kernel<<<(totH + blk - 1) / blk, blk, 0, stream>>>(cls_out_w, wClsH, 80, 128);
        int totR = 9 * 64 * IC;
        pack_reg_head_kernel<<<(totR + blk - 1) / blk, blk, 0, stream>>>(
            bbox_w, ctr_w, dim_w, ori_w, kp_w, dep_w, wRegH);
        pack_reg_bias_kernel<<<1, 32, 0, stream>>>(bbox_b, ctr_b, dim_b, ori_b, kp_b, dep_b, bReg);
        zero_kernel<<<1, 64, 0, stream>>>(zbuf, 64);
    }

    // --- per level, per tower ---
    for (int l = 0; l < 5; ++l) {
        const int H = LH[l], W = LW[l], HW = H * W;
        const int numPix = 2 * HW;
        const int lvlOff = LOFF[l];
        const int totE = numPix * IC;
        const dim3 cgridT((numPix + 31) / 32, 4);
        const dim3 cgridC((numPix + 31) / 32, 2);
        const dim3 cgridR((numPix + 31) / 32, 1);

        // cls tower
        pack_in_kernel<<<(totE + 255) / 256, 256, 0, stream>>>(feats[l], actA, HW, totE);
        _Float16* x = actA; _Float16* y = actB;
        for (int i = 0; i < 4; ++i) {
            conv3x3_wmma_kernel<0><<<cgridT, 256, 0, stream>>>(
                x, wClsT + (size_t)i * 9 * 256 * IC, cls_conv_b + i * 256, convO, zbuf,
                H, W, HW, numPix, 256, 256, 0, nullptr);
            gn_relu_kernel<<<2 * GROUPS, 256, 0, stream>>>(
                convO, cls_gn_w + i * 256, cls_gn_b + i * 256, y, HW);
            _Float16* t = x; x = y; y = t;
        }
        conv3x3_wmma_kernel<1><<<cgridC, 256, 0, stream>>>(
            x, wClsH, cls_out_b, nullptr, zbuf, H, W, HW, numPix, 128, 80, lvlOff, (float*)d_out);

        // reg tower
        pack_in_kernel<<<(totE + 255) / 256, 256, 0, stream>>>(feats[l], actA, HW, totE);
        x = actA; y = actB;
        for (int i = 0; i < 4; ++i) {
            conv3x3_wmma_kernel<0><<<cgridT, 256, 0, stream>>>(
                x, wRegT + (size_t)i * 9 * 256 * IC, reg_conv_b + i * 256, convO, zbuf,
                H, W, HW, numPix, 256, 256, 0, nullptr);
            gn_relu_kernel<<<2 * GROUPS, 256, 0, stream>>>(
                convO, reg_gn_w + i * 256, reg_gn_b + i * 256, y, HW);
            _Float16* t = x; x = y; y = t;
        }
        conv3x3_wmma_kernel<2><<<cgridR, 256, 0, stream>>>(
            x, wRegH, bReg, nullptr, zbuf, H, W, HW, numPix, 64, 26, lvlOff, (float*)d_out);
    }
}
